// LogitsProcessorWithPacked_15977278341385
// MI455X (gfx1250) — matrix operand order repaired and checked by reading, not compiled, not run on hardware
//
#include <hip/hip_runtime.h>
#include <hip/hip_bf16.h>
#include <cstdint>

// Problem constants from the reference.
#define T_TOK   512
#define H_DIM   2048
#define V_DIM   32000
#define N_DELTA 4

#define TILE_N  64              // vocab columns per workgroup
#define KB      32              // K depth per WMMA step
#define NMAT    5               // dense embedding + 4 adapter deltas
#define NKITER  (H_DIM / KB)    // 64
// LDS row stride in bf16 elements: 32 data + 8 pad = 80 bytes (16B aligned,
// and 20-dword stride -> lanes 0..15 land on distinct banks of the 64).
#define LDS_STRIDE 40
#define BUF_ELEMS  (NMAT * TILE_N * LDS_STRIDE)   // 12800 bf16 = 25600 B per buffer

typedef __attribute__((ext_vector_type(16))) __bf16 v16bf;
typedef __attribute__((ext_vector_type(8)))  __bf16 v8bf;
typedef __attribute__((ext_vector_type(4)))  __bf16 v4bf;
typedef __attribute__((ext_vector_type(8)))  float  v8f;
typedef __attribute__((ext_vector_type(4)))  float  f4;

// Pack 16 fp32 values (4x float4) into a 16-wide bf16 fragment register.
static __device__ __forceinline__ v16bf pack_bf16(f4 a, f4 b, f4 c, f4 d) {
  v16bf r;
  r[0]  = (__bf16)a.x; r[1]  = (__bf16)a.y; r[2]  = (__bf16)a.z; r[3]  = (__bf16)a.w;
  r[4]  = (__bf16)b.x; r[5]  = (__bf16)b.y; r[6]  = (__bf16)b.z; r[7]  = (__bf16)b.w;
  r[8]  = (__bf16)c.x; r[9]  = (__bf16)c.y; r[10] = (__bf16)c.z; r[11] = (__bf16)c.w;
  r[12] = (__bf16)d.x; r[13] = (__bf16)d.y; r[14] = (__bf16)d.z; r[15] = (__bf16)d.w;
  return r;
}

// Load a 32-byte bf16 fragment from LDS as two ds_load_b128.
static __device__ __forceinline__ v16bf ld_frag_lds(const __bf16* p) {
  v8bf lo = *(const v8bf*)(p);
  v8bf hi = *(const v8bf*)(p + 16);
  return __builtin_shufflevector(lo, hi, 0, 1, 2, 3, 4, 5, 6, 7,
                                 8, 9, 10, 11, 12, 13, 14, 15);
}

// Cooperatively stage the unique B panel (5 matrices x KB x TILE_N fp32) for
// one K step into LDS as bf16. 2560 float4 chunks / 256 threads = 10 each.
// Chunk c -> mat = c/512, n = (c%512)/8, kq = c%8  (coalesced in k, then n).
// B is use-once streaming data: load non-temporal so the 1.3 GB stream does
// not sweep hidden_states (L2-resident, re-read by every workgroup) out of
// the 192 MB L2.
static __device__ __forceinline__ void stage_B(
    __bf16* __restrict__ dst, const float* __restrict__ emb,
    const float* __restrict__ wstk, int n0, int kk, int tid) {
#pragma unroll
  for (int j = 0; j < 10; ++j) {
    const int c   = tid + j * 256;
    const int mat = c >> 9;        // / 512
    const int rem = c & 511;
    const int n   = rem >> 3;
    const int kq  = rem & 7;
    const float* Bm =
        (mat == 0) ? emb : (wstk + (size_t)(mat - 1) * (size_t)V_DIM * H_DIM);
    f4 x = __builtin_nontemporal_load(
        (const f4*)(Bm + (size_t)(n0 + n) * H_DIM + kk + kq * 4));
    v4bf o;
    o[0] = (__bf16)x.x; o[1] = (__bf16)x.y; o[2] = (__bf16)x.z; o[3] = (__bf16)x.w;
    *(v4bf*)(dst + (mat * TILE_N + n) * LDS_STRIDE + kq * 4) = o;
  }
}

// Fused dense vocab projection + per-token LoRA-delta GEMM.
// Grid: V/64 workgroups. Block: 256 threads = 8 wave32.
// Wave w computes tokens [64w, 64w+64) x vocab [64*bx, 64*bx+64).
__global__ __launch_bounds__(256, 1) void fused_logits_wmma(
    const float* __restrict__ hidden,   // [T, H] fp32
    const float* __restrict__ emb,      // [V, H] fp32
    const float* __restrict__ wstk,     // [D, V, H] fp32
    const long long* __restrict__ sidx, // [T] int64 adapter index (or -1)
    float* __restrict__ out)            // [T, V] fp32
{
  __shared__ __bf16 lbuf[2][BUF_ELEMS];   // double-buffered bf16 B panel

  const int tid   = threadIdx.x;
  const int lane  = tid & 31;
  const int wave  = tid >> 5;
  const int l15   = lane & 15;
  const int lhalf = lane >> 4;
  const int n0    = blockIdx.x * TILE_N;  // vocab panel base
  const int t0    = wave * 64;            // token panel base for this wave

  // Adapter index for the token row this lane owns in each M-frag.
  // (In the 16-bit A layout both lane halves map to rows M=0..15 = lane&15.)
  int adap[4];
#pragma unroll
  for (int mi = 0; mi < 4; ++mi)
    adap[mi] = (int)sidx[t0 + mi * 16 + l15];

  const v8f   vzero = {};
  const v16bf bzero = {};

  v8f acc[4][4];
#pragma unroll
  for (int mi = 0; mi < 4; ++mi)
#pragma unroll
    for (int nj = 0; nj < 4; ++nj)
      acc[mi][nj] = vzero;

  // Prologue: stage the first B panel.
  stage_B(lbuf[0], emb, wstk, n0, 0, tid);
  __syncthreads();

  for (int kt = 0; kt < NKITER; ++kt) {
    const int kk = kt * KB;
    const __bf16* cur = lbuf[kt & 1];

    // Stage next panel into the other buffer while we compute this one.
    // (Safe: the barrier at the end of the previous iteration guarantees all
    // reads of that buffer have completed.)
    if (kt + 1 < NKITER)
      stage_B(lbuf[(kt + 1) & 1], emb, wstk, n0, kk + KB, tid);

    // ---- A fragments direct from global (L2-resident, per-wave unique,
    // regular temporal hint so hidden stays hot in L2).
    // 16-bit A 16x32 layout: lanes 0-15 K 0..7 / 16..23, lanes 16-31 K 8..15 / 24..31.
    v16bf aF[4];
#pragma unroll
    for (int mi = 0; mi < 4; ++mi) {
      const float* p =
          hidden + (size_t)(t0 + mi * 16 + l15) * H_DIM + kk + lhalf * 8;
      f4 x0 = *(const f4*)(p);
      f4 x1 = *(const f4*)(p + 4);
      f4 x2 = *(const f4*)(p + 16);
      f4 x3 = *(const f4*)(p + 20);
      aF[mi] = pack_bf16(x0, x1, x2, x3);
    }

    // ---- 5 B matrices from LDS: m==0 dense embedding, m==1..4 adapter deltas.
#pragma unroll
    for (int m = 0; m < NMAT; ++m) {
      // B 32x16 layout: lane = column, lanes 0-15 K 0..7 / 16..23, lanes 16-31 K 8..15 / 24..31.
      v16bf bF[4];
#pragma unroll
      for (int nj = 0; nj < 4; ++nj) {
        const __bf16* p =
            cur + (m * TILE_N + nj * 16 + l15) * LDS_STRIDE + lhalf * 8;
        bF[nj] = ld_frag_lds(p);
      }

#pragma unroll
      for (int mi = 0; mi < 4; ++mi) {
        // Per-lane row mask for the adapter GEMMs (wave-uniform control flow;
        // EXEC stays all-ones as WMMA requires — mask is a VALU select).
        v16bf a = aF[mi];
        if (m > 0) a = (adap[mi] == (m - 1)) ? a : bzero;
#pragma unroll
        for (int nj = 0; nj < 4; ++nj) {
          acc[mi][nj] = __builtin_amdgcn_wmma_f32_16x16x32_bf16(
              /*neg_a=*/false, a, /*neg_b=*/false, bF[nj],
              /*c_mod=*/(short)0, acc[mi][nj],
              /*reuse_a=*/false, /*reuse_b=*/false);
        }
      }
    }

    // One barrier per K step closes both buffers' producer/consumer windows.
    __syncthreads();
  }

  // ---- Store C: VGPR r -> M = r + 8*(lane>=16), N = lane&15.
  // Output is write-once: non-temporal stores keep it from dirtying L2.
#pragma unroll
  for (int mi = 0; mi < 4; ++mi) {
#pragma unroll
    for (int nj = 0; nj < 4; ++nj) {
      const int col = n0 + nj * 16 + l15;
#pragma unroll
      for (int r = 0; r < 8; ++r) {
        const int row = t0 + mi * 16 + lhalf * 8 + r;
        __builtin_nontemporal_store(acc[mi][nj][r],
                                    out + (size_t)row * V_DIM + col);
      }
    }
  }
}

extern "C" void kernel_launch(void* const* d_in, const int* in_sizes, int n_in,
                              void* d_out, int out_size, void* d_ws, size_t ws_size,
                              hipStream_t stream) {
  const float*     hidden = (const float*)d_in[0];      // [T, H]
  const float*     emb    = (const float*)d_in[1];      // [V, H]
  const float*     wstk   = (const float*)d_in[2];      // [D, V, H]
  const long long* sidx   = (const long long*)d_in[3];  // [T] int64
  float*           out    = (float*)d_out;              // [T, V]

  dim3 grid(V_DIM / TILE_N);  // 500 workgroups, each owns a 64-wide vocab panel
  dim3 block(256);            // 8 wave32 -> 8 token slices of 64
  fused_logits_wmma<<<grid, block, 0, stream>>>(hidden, emb, wstk, sidx, out);
}
